// _LSTMModule_90323162235371
// MI455X (gfx1250) — matrix operand order repaired and checked by reading, not compiled, not run on hardware
//
#include <hip/hip_runtime.h>

typedef _Float16 v16h __attribute__((ext_vector_type(16)));
typedef _Float16 v8h  __attribute__((ext_vector_type(8)));
typedef float    v8f  __attribute__((ext_vector_type(8)));

#define HID   64
#define T_LEN 1024
#define BT    16      // batch tile per workgroup
#define NWAVE 4
#define NTHR  128

// --- activations: prefer hardware v_tanh_f32 (CDNA5 TRANS op), fallback to exp+rcp ---
#if __has_builtin(__builtin_amdgcn_tanhf)
#define TANHF(x) __builtin_amdgcn_tanhf(x)
#elif __has_builtin(__builtin_amdgcn_tanh_f32)
#define TANHF(x) __builtin_amdgcn_tanh_f32(x)
#else
__device__ __forceinline__ float tanh_fb(float x) {
    return 1.0f - 2.0f * __builtin_amdgcn_rcpf(1.0f + __expf(2.0f * x));
}
#define TANHF(x) tanh_fb(x)
#endif
// sigmoid(x) = 0.5 + 0.5*tanh(0.5x)  (mul + tanh + fma)
__device__ __forceinline__ float sigm(float x) { return 0.5f + 0.5f * TANHF(0.5f * x); }
// sigmoid(a+b) with pre-halved bias hb = 0.5*b: fma + tanh + fma
__device__ __forceinline__ float sigm_b(float a, float hb) { return 0.5f + 0.5f * TANHF(fmaf(0.5f, a, hb)); }

__device__ __forceinline__ v16h load_w16(const float* __restrict__ p) {
    v16h r;
#pragma unroll
    for (int j = 0; j < 16; ++j) r[j] = (_Float16)p[j + (j >= 8 ? 8 : 0)];
    return r;
}

__global__ __launch_bounds__(NTHR, 1) void lstm2_wmma_kernel(
    const float* __restrict__ x,
    const float* __restrict__ Wih0, const float* __restrict__ Whh0,
    const float* __restrict__ bih0, const float* __restrict__ bhh0,
    const float* __restrict__ Wih1, const float* __restrict__ Whh1,
    const float* __restrict__ bih1, const float* __restrict__ bhh1,
    const float* __restrict__ fc1w, const float* __restrict__ fc1b,
    const float* __restrict__ fc2w, const float* __restrict__ fc2b,
    float* __restrict__ out)
{
    // double-buffered hidden-state tiles (f16, A-operand staging) + x chunk
    __shared__ __attribute__((aligned(16))) _Float16 h0s[2][BT][HID];
    __shared__ __attribute__((aligned(16))) _Float16 h1s[2][BT][HID];
    __shared__ __attribute__((aligned(16))) float    xs[BT][64];

    const int ltid = threadIdx.x;
    const int wid  = ltid >> 5;     // wave 0..3 : owns hidden cols [16*wid, 16*wid+16)
    const int lane = ltid & 31;
    const int lo   = lane & 15;
    const int hi   = lane >> 4;
    const int base = blockIdx.x * BT;

    // ---- one-time setup: gate-column constants + weight B-fragments into VGPRs ----
    float wih0g[4], bias0g[4], bias1g[4], hbias1g[4];
    v16h b0[4][2], b1i[4][2], b1h[4][2];   // 192 VGPRs of resident weights
#pragma unroll
    for (int g = 0; g < 4; ++g) {
        const int gc = g * 64 + wid * 16 + lo;   // gate row in [0,256)
        wih0g[g]  = Wih0[gc];                    // INPUT_SIZE == 1
        bias0g[g] = bih0[gc] + bhh0[gc];
        bias1g[g] = bih1[gc] + bhh1[gc];
        hbias1g[g] = 0.5f * bias1g[g];
#pragma unroll
        for (int kt = 0; kt < 2; ++kt) {
            // B[K][N] = W[N][K]; lane holds N = gc, K = kt*32 + hi*8 + j (j<8), kt*32 + 16 + hi*8 + (j-8) (j>=8)
            const int off = gc * HID + kt * 32 + hi * 8;
            b0[g][kt]  = load_w16(Whh0 + off);
            b1i[g][kt] = load_w16(Wih1 + off);
            b1h[g][kt] = load_w16(Whh1 + off);
        }
    }

    for (int e = ltid; e < 2 * BT * HID; e += NTHR) {
        ((_Float16*)h0s)[e] = (_Float16)0.0f;
        ((_Float16*)h1s)[e] = (_Float16)0.0f;
    }

    float c0[8], c1[8];
#pragma unroll
    for (int v = 0; v < 8; ++v) { c0[v] = 0.0f; c1[v] = 0.0f; }

    const int ncol = wid * 16 + lo;   // hidden column this lane writes
    const int arow = lo;              // A-fragment M row this lane reads

    for (int t = 0; t < T_LEN; ++t) {
        const int wb = t & 1, rb = wb ^ 1;

        if ((t & 63) == 0) {          // refill x chunk (uniform branch, EXEC stays full)
            __syncthreads();
            for (int e = ltid; e < BT * 64; e += NTHR) {
                const int r = e >> 6, tt = e & 63;
                xs[r][tt] = x[(size_t)(base + r) * T_LEN + t + tt];
            }
            __syncthreads();
        }

        // ================= layer 0 =================
        v16h a0[2];
#pragma unroll
        for (int kt = 0; kt < 2; ++kt) {   // A 16x32 f16 layout: halfs 0-7 K=hi*8+j, 8-15 K=16+hi*8+j
            v8h al = *(const v8h*)&h0s[rb][arow][kt * 32 + hi * 8];
            v8h ah = *(const v8h*)&h0s[rb][arow][kt * 32 + 16 + hi * 8];
            a0[kt] = __builtin_shufflevector(al, ah, 0,1,2,3,4,5,6,7,8,9,10,11,12,13,14,15);
        }
        float xv[8];
#pragma unroll
        for (int v = 0; v < 8; ++v) xv[v] = xs[v + 8 * hi][t & 63];

        v8f acc[4];
#pragma unroll
        for (int g = 0; g < 4; ++g) {
#pragma unroll
            for (int v = 0; v < 8; ++v) acc[g][v] = xv[v] * wih0g[g] + bias0g[g];  // useful init work
            acc[g] = __builtin_amdgcn_wmma_f32_16x16x32_f16(false, a0[0], false, b0[g][0], (short)0, acc[g], false, false);
            acc[g] = __builtin_amdgcn_wmma_f32_16x16x32_f16(false, a0[1], false, b0[g][1], (short)0, acc[g], false, false);
        }
#pragma unroll
        for (int v = 0; v < 8; ++v) {       // C layout: element v -> M = v + 8*hi, N = lo
            const float ig = sigm(acc[0][v]);
            const float fg = sigm(acc[1][v]);
            const float gg = TANHF(acc[2][v]);
            const float og = sigm(acc[3][v]);
            const float c  = fg * c0[v] + ig * gg;
            c0[v] = c;
            h0s[wb][v + 8 * hi][ncol] = (_Float16)(og * TANHF(c));
        }
        __syncthreads();

        // ================= layer 1 =================
        v16h ah0[2], ah1[2];
#pragma unroll
        for (int kt = 0; kt < 2; ++kt) {
            v8h al0 = *(const v8h*)&h0s[wb][arow][kt * 32 + hi * 8];
            v8h ah0_ = *(const v8h*)&h0s[wb][arow][kt * 32 + 16 + hi * 8];
            ah0[kt] = __builtin_shufflevector(al0, ah0_, 0,1,2,3,4,5,6,7,8,9,10,11,12,13,14,15);
            v8h al1 = *(const v8h*)&h1s[rb][arow][kt * 32 + hi * 8];
            v8h ah1_ = *(const v8h*)&h1s[rb][arow][kt * 32 + 16 + hi * 8];
            ah1[kt] = __builtin_shufflevector(al1, ah1_, 0,1,2,3,4,5,6,7,8,9,10,11,12,13,14,15);
        }
        v8f acc1[4];
#pragma unroll
        for (int g = 0; g < 4; ++g) {
            v8f z = {};   // inline-0 C operand on first WMMA, no v_mov init
            z = __builtin_amdgcn_wmma_f32_16x16x32_f16(false, ah0[0], false, b1i[g][0], (short)0, z, false, false);
            z = __builtin_amdgcn_wmma_f32_16x16x32_f16(false, ah0[1], false, b1i[g][1], (short)0, z, false, false);
            z = __builtin_amdgcn_wmma_f32_16x16x32_f16(false, ah1[0], false, b1h[g][0], (short)0, z, false, false);
            z = __builtin_amdgcn_wmma_f32_16x16x32_f16(false, ah1[1], false, b1h[g][1], (short)0, z, false, false);
            acc1[g] = z;
        }
#pragma unroll
        for (int v = 0; v < 8; ++v) {       // bias folded into activation inputs
            const float ig = sigm_b(acc1[0][v], hbias1g[0]);
            const float fg = sigm_b(acc1[1][v], hbias1g[1]);
            const float gg = TANHF(acc1[2][v] + bias1g[2]);
            const float og = sigm_b(acc1[3][v], hbias1g[3]);
            const float c  = fg * c1[v] + ig * gg;
            c1[v] = c;
            h1s[wb][v + 8 * hi][ncol] = (_Float16)(og * TANHF(c));
        }
        __syncthreads();
    }

    // ---- FC head on last hidden state (h1s buffer (T-1)&1 == 1) ----
    if (ltid < BT) {
        const int r = ltid;
        float o2 = fc2b[0];
        for (int j = 0; j < 32; ++j) {
            float s = fc1b[j];
#pragma unroll
            for (int k = 0; k < HID; ++k) s += (float)h1s[1][r][k] * fc1w[j * HID + k];
            o2 += fmaxf(s, 0.0f) * fc2w[j];
        }
        out[base + r] = o2;
    }
}

extern "C" void kernel_launch(void* const* d_in, const int* in_sizes, int n_in,
                              void* d_out, int out_size, void* d_ws, size_t ws_size,
                              hipStream_t stream)
{
    const float* x    = (const float*)d_in[0];
    const float* Wih0 = (const float*)d_in[1];
    const float* Whh0 = (const float*)d_in[2];
    const float* bih0 = (const float*)d_in[3];
    const float* bhh0 = (const float*)d_in[4];
    const float* Wih1 = (const float*)d_in[5];
    const float* Whh1 = (const float*)d_in[6];
    const float* bih1 = (const float*)d_in[7];
    const float* bhh1 = (const float*)d_in[8];
    const float* fc1w = (const float*)d_in[9];
    const float* fc1b = (const float*)d_in[10];
    const float* fc2w = (const float*)d_in[11];
    const float* fc2b = (const float*)d_in[12];
    float* out = (float*)d_out;

    lstm2_wmma_kernel<<<dim3(1024 / BT), dim3(NTHR), 0, stream>>>(
        x, Wih0, Whh0, bih0, bhh0, Wih1, Whh1, bih1, bhh1,
        fc1w, fc1b, fc2w, fc2b, out);
}
